// Block_10024453669245
// MI455X (gfx1250) — matrix operand order repaired and checked by reading, not compiled, not run on hardware
//
#include <hip/hip_runtime.h>
#include <math.h>

// ---------------------------------------------------------------------------
// Transformer block (pre-LN attention w/o out-proj + pre-LN MLP), MI455X.
// B=2, T=2048, C=2048, H=16, D=128.
//  - all matmuls: v_wmma_f32_16x16x32_bf16 (f32 accumulate)
//  - GEMM tiles staged to LDS by the Tensor Data Mover (tensor_load_to_lds,
//    TENSORcnt, double-buffered, 128x64 tiles)
//  - attention K and V tiles staged with global_load_async_to_lds_b128
//  - weights transposed+cast to bf16 once; V transposed to [B,H,D,T] once,
//    so every LDS tile everywhere is a plain contiguous/strided row copy
// Workspace: ~293 MB.
// ---------------------------------------------------------------------------

typedef __attribute__((ext_vector_type(16))) __bf16 v16bf;
typedef __attribute__((ext_vector_type(8)))  __bf16 bf16x8;
typedef __attribute__((ext_vector_type(4)))  __bf16 bf16x4;
typedef __attribute__((ext_vector_type(8)))  float  v8f;
typedef __attribute__((ext_vector_type(4)))  unsigned v4u;
typedef __attribute__((ext_vector_type(8)))  int      v8i;
typedef __attribute__((ext_vector_type(4)))  int      v4i;

#define B_DIM 2
#define T_DIM 2048
#define C_DIM 2048
#define H_DIM 16
#define D_DIM 128
#define M_ROWS (B_DIM * T_DIM)   // 4096

static __device__ __forceinline__ v8f wmma_bf16(v16bf a, v16bf b, v8f c) {
  return __builtin_amdgcn_wmma_f32_16x16x32_bf16(false, a, false, b,
                                                 (short)0, c, false, false);
}

static __device__ __forceinline__ v16bf mk16(const __bf16* lo, const __bf16* hi8) {
  bf16x8 a = *(const bf16x8*)lo;
  bf16x8 b = *(const bf16x8*)hi8;
  v16bf r;
#pragma unroll
  for (int i = 0; i < 8; ++i) { r[i] = a[i]; r[i + 8] = b[i]; }
  return r;
}

static __device__ __forceinline__ float gelu_f(float x) {
  const float c = 0.7978845608028654f; // sqrt(2/pi)
  float x3 = x * x * x;
  return 0.5f * x * (1.0f + tanhf(c * (x + 0.044715f * x3)));
}

// --- Tensor Data Mover: load a 128-row x 64-half tile (row stride K halfs)
// from global bf16 into LDS.  D# layout per CDNA5 ISA ch.8 (2D tensor).
// 6-arg clang-23 builtin: (v4u, v8i, v4i, v4i, v8i, cpol).
static __device__ __forceinline__ void tdm_load_tile_128x64(
    unsigned lds_addr, const __bf16* gptr, int K) {
  unsigned long long ga = (unsigned long long)(uintptr_t)gptr;
  v4u g0;
  g0[0] = 1u;                                   // count=1 (valid user D#)
  g0[1] = lds_addr;                             // lds_addr
  g0[2] = (unsigned)(ga & 0xFFFFFFFFu);         // global_addr lo
  g0[3] = (unsigned)((ga >> 32) & 0x01FFFFFFu)  // global_addr hi (57-bit)
        | (2u << 30);                           // type=2 ("image")
  v8i g1;
  g1[0] = 0x00010000;                           // data_size=1 (2B), mask=0
  g1[1] = (K & 0xFFFF) << 16;                   // tensor_dim0[15:0]
  g1[2] = ((K >> 16) & 0xFFFF) | (128 << 16);   // dim0 hi | tensor_dim1 lo
  g1[3] = (64 << 16);                           // tile_dim0 = 64
  g1[4] = 128;                                  // tile_dim1 = 128
  g1[5] = K;                                    // tensor_dim0_stride lo
  g1[6] = 0;
  g1[7] = 0;
  v4i z4; z4[0] = 0; z4[1] = 0; z4[2] = 0; z4[3] = 0;
  v8i z8;
#pragma unroll
  for (int i = 0; i < 8; ++i) z8[i] = 0;
  __builtin_amdgcn_tensor_load_to_lds(g0, g1, z4, z4, z8, 0);
}

// --- Async global->LDS 16B copy (per lane), tracked on ASYNCcnt.
static __device__ __forceinline__ void async_copy16(unsigned lds_addr,
                                                    const __bf16* g) {
  unsigned long long ga = (unsigned long long)(uintptr_t)g;
  asm volatile("global_load_async_to_lds_b128 %0, %1, off"
               :: "v"(lds_addr), "v"(ga) : "memory");
}
static __device__ __forceinline__ void wait_async0() {
  asm volatile("s_wait_asynccnt 0x0" ::: "memory");
}

// ---------------------------------------------------------------------------
// Transpose + cast: in f32 [K][N] -> out bf16 [N][K].  32x32 tiles.
// ---------------------------------------------------------------------------
__global__ __launch_bounds__(256)
void transpose_cast_bf16(const float* __restrict__ in, __bf16* __restrict__ out,
                         int K, int N) {
  __shared__ float tile[32][33];
  const int k0 = blockIdx.y * 32, n0 = blockIdx.x * 32;
  const int t = threadIdx.x;
  const int r = t >> 3, c4 = (t & 7) * 4;
  float4 v = *(const float4*)(in + (size_t)(k0 + r) * N + n0 + c4);
  tile[r][c4 + 0] = v.x; tile[r][c4 + 1] = v.y;
  tile[r][c4 + 2] = v.z; tile[r][c4 + 3] = v.w;
  __syncthreads();
  bf16x4 o;
  o[0] = (__bf16)tile[c4 + 0][r]; o[1] = (__bf16)tile[c4 + 1][r];
  o[2] = (__bf16)tile[c4 + 2][r]; o[3] = (__bf16)tile[c4 + 3][r];
  *(bf16x4*)(out + (size_t)(n0 + r) * K + k0 + c4) = o;
}

// ---------------------------------------------------------------------------
// V transpose: qkv [B,T,3C] (V slice, per head [T][D]) -> vt [B*H][D][T] bf16.
// 32x32 tiles via LDS.  grid: x = T/32, y = (B*H)*(D/32).
// ---------------------------------------------------------------------------
__global__ __launch_bounds__(256)
void transpose_v_bf16(const __bf16* __restrict__ qkv, __bf16* __restrict__ vt) {
  __shared__ __bf16 tile[32][33];
  const int t0 = blockIdx.x * 32;
  const int bh = blockIdx.y >> 2, d0 = (blockIdx.y & 3) * 32;
  const int b = bh >> 4, h = bh & 15;
  const int tt = threadIdx.x;
  const int r = tt >> 3, c4 = (tt & 7) * 4;
  const __bf16* src = qkv + (size_t)(b * T_DIM + t0 + r) * (3 * C_DIM) +
                      2 * C_DIM + h * D_DIM + d0 + c4;
  bf16x4 v = *(const bf16x4*)src;
  tile[r][c4 + 0] = v[0]; tile[r][c4 + 1] = v[1];
  tile[r][c4 + 2] = v[2]; tile[r][c4 + 3] = v[3];
  __syncthreads();
  bf16x4 o;
  o[0] = tile[c4 + 0][r]; o[1] = tile[c4 + 1][r];
  o[2] = tile[c4 + 2][r]; o[3] = tile[c4 + 3][r];
  *(bf16x4*)(vt + (size_t)(bh * D_DIM + d0 + r) * T_DIM + t0 + c4) = o;
}

// ---------------------------------------------------------------------------
// LayerNorm (weight only, eps=1e-5) over C=2048, one block per row -> bf16
// ---------------------------------------------------------------------------
__global__ __launch_bounds__(256)
void layernorm_bf16(const float* __restrict__ x, const float* __restrict__ w,
                    __bf16* __restrict__ out) {
  const int row = blockIdx.x;
  const int tid = threadIdx.x;
  const float* xr = x + (size_t)row * C_DIM;
  __shared__ float red[256];

  float vals[8];
  float s = 0.f;
#pragma unroll
  for (int j = 0; j < 8; ++j) { vals[j] = xr[tid + j * 256]; s += vals[j]; }
  red[tid] = s; __syncthreads();
  for (int off = 128; off > 0; off >>= 1) {
    if (tid < off) red[tid] += red[tid + off];
    __syncthreads();
  }
  const float mean = red[0] * (1.0f / C_DIM);
  __syncthreads();

  float vs = 0.f;
#pragma unroll
  for (int j = 0; j < 8; ++j) { float d = vals[j] - mean; vs += d * d; }
  red[tid] = vs; __syncthreads();
  for (int off = 128; off > 0; off >>= 1) {
    if (tid < off) red[tid] += red[tid + off];
    __syncthreads();
  }
  const float rstd = rsqrtf(red[0] * (1.0f / C_DIM) + 1e-5f);

#pragma unroll
  for (int j = 0; j < 8; ++j) {
    int c = tid + j * 256;
    out[(size_t)row * C_DIM + c] = (__bf16)((vals[j] - mean) * rstd * w[c]);
  }
}

// ---------------------------------------------------------------------------
// bf16 GEMM: Out[M,N] = A[M,K] * Bt[N,K]^T.  128x128 block tile, 8 waves
// (4Mx2N), wave tile 32x64 = 2x4 WMMA frags.  K tile = 64 (two 32-deep WMMA
// sub-steps per sync).  LDS tiles fetched by the TDM, double-buffered.
// EPI: 0 = store bf16, 1 = store gelu(x) bf16, 2 = store f32 + residual
// ---------------------------------------------------------------------------
template <int EPI>
__global__ __launch_bounds__(256)
void gemm_bf16(const __bf16* __restrict__ A, const __bf16* __restrict__ Bt,
               void* __restrict__ Out, const float* __restrict__ Res,
               int M, int N, int K) {
  __shared__ alignas(128) __bf16 As[2][128][64];   // [m][k]
  __shared__ alignas(128) __bf16 Bs[2][128][64];   // [n][k]

  const int tid = threadIdx.x;
  const int lane = tid & 31;
  const int wid = tid >> 5;
  const int waveM = wid & 3, waveN = wid >> 2;
  const int ln = lane & 15, hi = lane >> 4;
  const int m0 = blockIdx.y * 128, n0 = blockIdx.x * 128;

  const __bf16* Abase = A + (size_t)m0 * K;
  const __bf16* Bbase = Bt + (size_t)n0 * K;

  v8f acc[2][4];
#pragma unroll
  for (int i = 0; i < 2; ++i)
#pragma unroll
    for (int j = 0; j < 4; ++j)
#pragma unroll
      for (int r = 0; r < 8; ++r) acc[i][j][r] = 0.f;

  const int nIter = K / 64;
  if (wid == 0)
    tdm_load_tile_128x64((unsigned)(uintptr_t)&As[0][0][0], Abase, K);
  else if (wid == 1)
    tdm_load_tile_128x64((unsigned)(uintptr_t)&Bs[0][0][0], Bbase, K);

  for (int it = 0; it < nIter; ++it) {
    const int buf = it & 1;
    if (wid < 2) __builtin_amdgcn_s_wait_tensorcnt(0);  // tile(it) landed
    __syncthreads();
    if (it + 1 < nIter) {                // kick off tile(it+1) -> other buffer
      const int nb = buf ^ 1;
      const int k0n = (it + 1) * 64;
      if (wid == 0)
        tdm_load_tile_128x64((unsigned)(uintptr_t)&As[nb][0][0], Abase + k0n, K);
      else if (wid == 1)
        tdm_load_tile_128x64((unsigned)(uintptr_t)&Bs[nb][0][0], Bbase + k0n, K);
    }

#pragma unroll
    for (int ks = 0; ks < 2; ++ks) {
      v16bf af[2], bfrag[4];
#pragma unroll
      for (int mf = 0; mf < 2; ++mf) {
        int m = waveM * 32 + mf * 16 + ln;
        af[mf] = mk16(&As[buf][m][ks * 32 + hi * 8],
                      &As[buf][m][ks * 32 + 16 + hi * 8]);
      }
#pragma unroll
      for (int nf = 0; nf < 4; ++nf) {
        int n = waveN * 64 + nf * 16 + ln;
        bfrag[nf] = mk16(&Bs[buf][n][ks * 32 + hi * 8],
                         &Bs[buf][n][ks * 32 + 16 + hi * 8]);
      }
#pragma unroll
      for (int mf = 0; mf < 2; ++mf)
#pragma unroll
        for (int nf = 0; nf < 4; ++nf)
          acc[mf][nf] = wmma_bf16(af[mf], bfrag[nf], acc[mf][nf]);
    }
  }

  // Epilogue: C/D layout -> element r is (m = r + 8*hi, n = lane%16)
#pragma unroll
  for (int mf = 0; mf < 2; ++mf)
#pragma unroll
    for (int nf = 0; nf < 4; ++nf)
#pragma unroll
      for (int r = 0; r < 8; ++r) {
        int gm = m0 + waveM * 32 + mf * 16 + r + hi * 8;
        int gn = n0 + waveN * 64 + nf * 16 + ln;
        size_t idx = (size_t)gm * N + gn;
        float v = acc[mf][nf][r];
        if (EPI == 0) {
          ((__bf16*)Out)[idx] = (__bf16)v;
        } else if (EPI == 1) {
          ((__bf16*)Out)[idx] = (__bf16)gelu_f(v);
        } else {
          ((float*)Out)[idx] = v + Res[idx];
        }
      }
}

// ---------------------------------------------------------------------------
// Flash attention (causal) + residual: x1 = x + softmax(QK^T/sqrt(D)) V.
// Block = 4 waves, 64 query rows; wave owns 16 rows; key blocks of 32;
// online softmax; K and V tiles staged with async global->LDS copies.
// ---------------------------------------------------------------------------
__global__ __launch_bounds__(128)
void attn_kernel(const __bf16* __restrict__ qkv, const __bf16* __restrict__ vt,
                 const float* __restrict__ x, float* __restrict__ x1) {
  const int qb = blockIdx.x;          // 0..T/64-1
  const int bh = blockIdx.y;          // 0..B*H-1
  const int b = bh >> 4, h = bh & 15;
  const int tid = threadIdx.x;
  const int lane = tid & 31, w = tid >> 5;
  const int ln = lane & 15, hi = lane >> 4;
  const int q0 = qb * 64 + w * 16;

  __shared__ alignas(64) __bf16 Ks[32][128];     // [key][d]
  __shared__ alignas(64) __bf16 Vs[128][32];     // [d][key]
  __shared__ alignas(64) __bf16 Ps[4][16][32];   // per-wave P tile [m][key]

  const size_t rowstride = 3 * C_DIM;

  // Q fragments: 16 rows x 128 d as 4 chunks of K=32
  v16bf qf[4];
  {
    const __bf16* qrow =
        qkv + (size_t)(b * T_DIM + q0 + ln) * rowstride + h * D_DIM;
#pragma unroll
    for (int c = 0; c < 4; ++c)
      qf[c] = mk16(qrow + c * 32 + hi * 8, qrow + c * 32 + 16 + hi * 8);
  }

  const float scale = 0.08838834764831845f;  // 1/sqrt(128)
  float mrow[8], lrow[8];
  v8f o[8];
#pragma unroll
  for (int r = 0; r < 8; ++r) { mrow[r] = -1e30f; lrow[r] = 0.f; }
#pragma unroll
  for (int dt = 0; dt < 8; ++dt)
#pragma unroll
    for (int r = 0; r < 8; ++r) o[dt][r] = 0.f;

  const int nkb = qb * 2 + 2;
  for (int kb = 0; kb < nkb; ++kb) {
    const int key0 = kb * 32;
    __syncthreads();
    // K tile: thread -> one key row quarter (4x16B async)
    {
      int kk = tid >> 2, c4 = tid & 3;
      const __bf16* src = qkv + (size_t)(b * T_DIM + key0 + kk) * rowstride +
                          C_DIM + h * D_DIM + c4 * 32;
#pragma unroll
      for (int j = 0; j < 4; ++j)
        async_copy16((unsigned)(uintptr_t)&Ks[kk][c4 * 32 + j * 8],
                     src + j * 8);
    }
    // V tile from pre-transposed vt: thread -> one d row (4x16B async)
    {
      const __bf16* src = vt + (size_t)(bh * D_DIM + tid) * T_DIM + key0;
#pragma unroll
      for (int j = 0; j < 4; ++j)
        async_copy16((unsigned)(uintptr_t)&Vs[tid][j * 8], src + j * 8);
    }
    wait_async0();
    __syncthreads();

    if (key0 <= q0 + 15) {  // wave-uniform causal skip
      v8f s[2];
#pragma unroll
      for (int nt = 0; nt < 2; ++nt) {
#pragma unroll
        for (int r = 0; r < 8; ++r) s[nt][r] = 0.f;
#pragma unroll
        for (int c = 0; c < 4; ++c) {
          v16bf kf = mk16(&Ks[nt * 16 + ln][c * 32 + hi * 8],
                          &Ks[nt * 16 + ln][c * 32 + 16 + hi * 8]);
          s[nt] = wmma_bf16(qf[c], kf, s[nt]);
        }
      }
#pragma unroll
      for (int r = 0; r < 8; ++r) {
        const int qr = q0 + r + hi * 8;
        float s0 = s[0][r] * scale, s1 = s[1][r] * scale;
        if (key0 + ln > qr)      s0 = -1e30f;
        if (key0 + 16 + ln > qr) s1 = -1e30f;
        float cur = fmaxf(s0, s1);
#pragma unroll
        for (int off = 1; off < 16; off <<= 1)
          cur = fmaxf(cur, __shfl_xor(cur, off, 32));
        float mnew = fmaxf(mrow[r], cur);
        float p0 = __expf(s0 - mnew), p1 = __expf(s1 - mnew);
        float ps = p0 + p1;
#pragma unroll
        for (int off = 1; off < 16; off <<= 1) ps += __shfl_xor(ps, off, 32);
        float alpha = __expf(mrow[r] - mnew);
        lrow[r] = lrow[r] * alpha + ps;
        mrow[r] = mnew;
#pragma unroll
        for (int dt = 0; dt < 8; ++dt) o[dt][r] *= alpha;
        Ps[w][r + hi * 8][ln]      = (__bf16)p0;
        Ps[w][r + hi * 8][16 + ln] = (__bf16)p1;
      }
      v16bf pf = mk16(&Ps[w][ln][hi * 8], &Ps[w][ln][16 + hi * 8]);
#pragma unroll
      for (int dt = 0; dt < 8; ++dt) {
        int d = dt * 16 + ln;
        v16bf vf = mk16(&Vs[d][hi * 8], &Vs[d][16 + hi * 8]);
        o[dt] = wmma_bf16(pf, vf, o[dt]);
      }
    }
  }

#pragma unroll
  for (int dt = 0; dt < 8; ++dt)
#pragma unroll
    for (int r = 0; r < 8; ++r) {
      int q = q0 + r + hi * 8;
      int c = h * D_DIM + dt * 16 + ln;
      size_t idx = (size_t)(b * T_DIM + q) * C_DIM + c;
      x1[idx] = x[idx] + o[dt][r] / lrow[r];
    }
}

// ---------------------------------------------------------------------------
// Host-side launch
// ---------------------------------------------------------------------------
extern "C" void kernel_launch(void* const* d_in, const int* in_sizes, int n_in,
                              void* d_out, int out_size, void* d_ws,
                              size_t ws_size, hipStream_t stream) {
  (void)in_sizes; (void)n_in; (void)out_size; (void)ws_size;
  const float* x     = (const float*)d_in[0];
  const float* w_qkv = (const float*)d_in[1];
  const float* w1    = (const float*)d_in[2];
  const float* w2    = (const float*)d_in[3];
  const float* ln_w  = (const float*)d_in[4];
  float* out = (float*)d_out;

  char* ws = (char*)d_ws;
  size_t off = 0;
  auto alloc = [&](size_t bytes) -> void* {
    void* p = ws + off;
    off += (bytes + 255) & ~(size_t)255;
    return p;
  };
  __bf16* h_bf    = (__bf16*)alloc((size_t)M_ROWS * C_DIM * 2);       // 16 MB
  __bf16* h2_bf   = (__bf16*)alloc((size_t)M_ROWS * C_DIM * 2);       // 16 MB
  __bf16* wqkv_t  = (__bf16*)alloc((size_t)C_DIM * 3 * C_DIM * 2);    // 24 MB [6144][2048]
  __bf16* w1_t    = (__bf16*)alloc((size_t)C_DIM * 4 * C_DIM * 2);    // 32 MB [8192][2048]
  __bf16* w2_t    = (__bf16*)alloc((size_t)4 * C_DIM * C_DIM * 2);    // 32 MB [2048][8192]
  __bf16* qkv_bf  = (__bf16*)alloc((size_t)M_ROWS * 3 * C_DIM * 2);   // 48 MB
  __bf16* vt_bf   = (__bf16*)alloc((size_t)B_DIM * H_DIM * D_DIM * T_DIM * 2); // 16 MB
  __bf16* mlp1_bf = (__bf16*)alloc((size_t)M_ROWS * 4 * C_DIM * 2);   // 64 MB
  float*  x1      = (float*)alloc((size_t)M_ROWS * C_DIM * 4);        // 32 MB

  // 1) weight transpose+cast f32 [K][N] -> bf16 [N][K]
  transpose_cast_bf16<<<dim3(3 * C_DIM / 32, C_DIM / 32), 256, 0, stream>>>(
      w_qkv, wqkv_t, C_DIM, 3 * C_DIM);
  transpose_cast_bf16<<<dim3(4 * C_DIM / 32, C_DIM / 32), 256, 0, stream>>>(
      w1, w1_t, C_DIM, 4 * C_DIM);
  transpose_cast_bf16<<<dim3(C_DIM / 32, 4 * C_DIM / 32), 256, 0, stream>>>(
      w2, w2_t, 4 * C_DIM, C_DIM);

  // 2) h = LN(x)
  layernorm_bf16<<<M_ROWS, 256, 0, stream>>>(x, ln_w, h_bf);

  // 3) qkv = h @ w_qkv
  gemm_bf16<0><<<dim3(3 * C_DIM / 128, M_ROWS / 128), 256, 0, stream>>>(
      h_bf, wqkv_t, (void*)qkv_bf, nullptr, M_ROWS, 3 * C_DIM, C_DIM);

  // 4) V -> [B*H][D][T]
  transpose_v_bf16<<<dim3(T_DIM / 32, B_DIM * H_DIM * (D_DIM / 32)), 256, 0,
                     stream>>>(qkv_bf, vt_bf);

  // 5) x1 = x + attention(qkv)
  attn_kernel<<<dim3(T_DIM / 64, B_DIM * H_DIM), 128, 0, stream>>>(
      qkv_bf, vt_bf, x, x1);

  // 6) h2 = LN(x1)
  layernorm_bf16<<<M_ROWS, 256, 0, stream>>>(x1, ln_w, h2_bf);

  // 7) mlp1 = gelu(h2 @ w1)
  gemm_bf16<1><<<dim3(4 * C_DIM / 128, M_ROWS / 128), 256, 0, stream>>>(
      h2_bf, w1_t, (void*)mlp1_bf, nullptr, M_ROWS, 4 * C_DIM, C_DIM);

  // 8) out = x1 + mlp1 @ w2
  gemm_bf16<2><<<dim3(C_DIM / 128, M_ROWS / 128), 256, 0, stream>>>(
      mlp1_bf, w2_t, (void*)out, x1, M_ROWS, C_DIM, 4 * C_DIM);
}
